// Strawberry_23665269801478
// MI455X (gfx1250) — matrix operand order repaired and checked by reading, not compiled, not run on hardware
//
#include <hip/hip_runtime.h>
#include <hip/hip_bf16.h>

// ---------------- problem constants ----------------
#define N_NODES   20000
#define N_EDGES   320000
#define HID       64
#define NFEAT     256        // 4*HID
#define L_LAYERS  3
#define N_GRAPHS  128

// ---------------- WMMA types ----------------
typedef __attribute__((ext_vector_type(16))) __bf16 v16bf;
typedef __attribute__((ext_vector_type(8)))  float  v8f;
typedef __attribute__((ext_vector_type(8)))  __bf16 bf16x8;

union AFrag { v16bf v; bf16x8 h[2]; };

static __device__ __forceinline__ __bf16 f2bf(float f) {
  union { float f; unsigned u; } in; in.f = f;
  unsigned u = in.u;
  u += 0x7fffu + ((u >> 16) & 1u);                 // round-to-nearest-even
  unsigned short hs = (unsigned short)(u >> 16);
  union { unsigned short s; __bf16 b; } out; out.s = hs;
  return out.b;
}

static __device__ __forceinline__ float silu(float x) {
  return x * (1.0f / (1.0f + __expf(-x)));
}

// ---------------- weight prep: src[K][N] f32 -> dst[N][K] bf16 ----------------
__global__ void straw_transpose_bf16(const float* __restrict__ src,
                                     __bf16* __restrict__ dst, int K, int N) {
  long idx = (long)blockIdx.x * blockDim.x + threadIdx.x;
  long total = (long)K * N;
  if (idx < total) {
    int k = (int)(idx / N);
    int n = (int)(idx % N);
    dst[(long)n * K + k] = f2bf(src[idx]);
  }
}

__global__ void straw_w1last(const float* __restrict__ msg_w1, float* __restrict__ w1last) {
  int i = blockIdx.x * blockDim.x + threadIdx.x;   // l*256 + n
  if (i < L_LAYERS * NFEAT) {
    int l = i >> 8, n = i & 255;
    w1last[i] = msg_w1[(long)l * 513 * NFEAT + 512 * NFEAT + n];
  }
}

// ---------------- node init: x = emb[z], v = broadcast(pos) ----------------
__global__ __launch_bounds__(256) void straw_init_nodes(
    const int* __restrict__ z, const float* __restrict__ pos,
    const float* __restrict__ emb, float* __restrict__ x_buf, float* __restrict__ v_buf) {
  long n = blockIdx.x; int t = threadIdx.x;
  if (t < HID)     x_buf[n * HID + t] = emb[(long)z[n] * HID + t];
  if (t < 3 * HID) v_buf[n * 3 * HID + t] = pos[n * 3 + t / HID];
}

// ---------------- per-layer prep: bf16 feats, pseudo, zero agg ----------------
__global__ __launch_bounds__(256) void straw_node_prep(
    const float* __restrict__ x_buf, const float* __restrict__ v_buf,
    __bf16* __restrict__ nodefeat, float* __restrict__ pseudo,
    float* __restrict__ agg_s, float* __restrict__ agg_v) {
  long n = blockIdx.x; int t = threadIdx.x;
  float val = (t < HID) ? x_buf[n * HID + t] : v_buf[n * 3 * HID + (t - HID)];
  nodefeat[n * NFEAT + t] = f2bf(val);
  if (t < HID) agg_s[n * HID + t] = 0.0f;
  else         agg_v[n * 3 * HID + (t - HID)] = 0.0f;
  if (t < 3) {
    float s = 0.0f;
    for (int h = 0; h < HID; ++h) s += v_buf[n * 3 * HID + t * HID + h];
    pseudo[n * 3 + t] = s * (1.0f / 64.0f);
  }
}

// ---------------- big edge-MLP kernel: 16 edges / block, 16 waves ----------------
__global__ __launch_bounds__(512) void straw_edge_mlp(
    const __bf16* __restrict__ nodefeat,   // [N][256]
    const float*  __restrict__ pseudo,     // [N][3]
    const int*    __restrict__ row, const int* __restrict__ col,
    const __bf16* __restrict__ w1t,        // [256][512] (n-major)
    const float*  __restrict__ w1last,     // [256]
    const float*  __restrict__ b1,         // [256]
    const __bf16* __restrict__ w2t,        // [256][256]
    const float*  __restrict__ b2,         // [256]
    float* __restrict__ agg_s,             // [N][64]
    float* __restrict__ agg_v)             // [N][192]
{
  __shared__ __align__(16) __bf16 sA1[16 * 512];
  __shared__ __align__(16) __bf16 sA2[16 * 256];
  __shared__ float sDist[16];
  __shared__ int   sRow[16];

  const int tid  = threadIdx.x;
  const int lane = tid & 31;
  const int wv   = tid >> 5;
  const long e0  = (long)blockIdx.x * 16;

  if (tid < 16) {
    int e = (int)e0 + tid;
    int r = row[e], c = col[e];
    sRow[tid] = r;
    float dx = pseudo[r*3+0] - pseudo[c*3+0];
    float dy = pseudo[r*3+1] - pseudo[c*3+1];
    float dz = pseudo[r*3+2] - pseudo[c*3+2];
    sDist[tid] = sqrtf(dx*dx + dy*dy + dz*dz);
  }
  // stage A: 16 rows x 512 bf16 = 1024 x 16B chunks
  #pragma unroll
  for (int c = tid; c < 1024; c += 512) {
    int m = c >> 6, part = c & 63;
    int e = (int)e0 + m;
    const __bf16* src = (part < 32)
        ? nodefeat + (long)row[e] * NFEAT + part * 8
        : nodefeat + (long)col[e] * NFEAT + (part - 32) * 8;
    *(bf16x8*)(&sA1[m * 512 + part * 8]) = *(const bf16x8*)src;
  }
  __syncthreads();

  const int nb   = wv * 16;
  const int ncol = nb + (lane & 15);
  const int mh   = (lane < 16) ? 0 : 8;
  const int m    = lane & 15;
  const int kaOff = (lane < 16) ? 0 : 8;
  const int kbOff = (lane < 16) ? 0 : 8;

  // GEMM1: 16x512 @ 512x256 tile, K-steps of 32
  v8f acc = {};
  for (int kk = 0; kk < 16; ++kk) {
    AFrag a, b;
    int ka = kk * 32 + kaOff;
    a.h[0] = *(const bf16x8*)(&sA1[m * 512 + ka]);
    a.h[1] = *(const bf16x8*)(&sA1[m * 512 + ka + 16]);
    const __bf16* bp = w1t + (long)ncol * 512 + kk * 32 + kbOff;
    b.h[0] = *(const bf16x8*)(bp);
    b.h[1] = *(const bf16x8*)(bp + 16);
    acc = __builtin_amdgcn_wmma_f32_16x16x32_bf16(false, a.v, false, b.v,
                                                  (short)0, acc, false, false);
  }
  {
    float bias = b1[ncol];
    float wl   = w1last[ncol];
    #pragma unroll
    for (int r = 0; r < 8; ++r) {
      int mm = mh + r;
      float val = acc[r] + bias + sDist[mm] * wl;
      sA2[mm * 256 + ncol] = f2bf(silu(val));
    }
  }
  __syncthreads();

  // GEMM2: 16x256 @ 256x256 tile
  v8f acc2 = {};
  for (int kk = 0; kk < 8; ++kk) {
    AFrag a, b;
    int ka = kk * 32 + kaOff;
    a.h[0] = *(const bf16x8*)(&sA2[m * 256 + ka]);
    a.h[1] = *(const bf16x8*)(&sA2[m * 256 + ka + 16]);
    const __bf16* bp = w2t + (long)ncol * 256 + kk * 32 + kbOff;
    b.h[0] = *(const bf16x8*)(bp);
    b.h[1] = *(const bf16x8*)(bp + 16);
    acc2 = __builtin_amdgcn_wmma_f32_16x16x32_bf16(false, a.v, false, b.v,
                                                   (short)0, acc2, false, false);
  }
  {
    float bias2 = b2[ncol];
    #pragma unroll
    for (int r = 0; r < 8; ++r) {
      int mm = mh + r;
      float val = silu(acc2[r] + bias2);
      int node = sRow[mm];
      if (ncol < HID) atomicAdd(&agg_s[(long)node * HID + ncol], val);
      else            atomicAdd(&agg_v[(long)node * 3 * HID + (ncol - HID)], val);
    }
  }
}

// ---------------- node update: 16 nodes / block, 4 waves ----------------
__global__ __launch_bounds__(128) void straw_node_update(
    float* __restrict__ x_buf, float* __restrict__ v_buf,     // in-place
    const float* __restrict__ agg_s, const float* __restrict__ agg_v,
    const __bf16* __restrict__ uw1t,  // [64][128]
    const float*  __restrict__ ub1,
    const __bf16* __restrict__ uw2t,  // [64][64]
    const float*  __restrict__ ub2,
    const __bf16* __restrict__ mixt,  // [64][64]
    float* __restrict__ xs_out, float* __restrict__ vs_out)
{
  __shared__ __align__(16) __bf16 sCat[16 * 128];
  __shared__ __align__(16) __bf16 sU[16 * 64];
  __shared__ __align__(16) __bf16 sV[16 * 192];

  const int tid  = threadIdx.x;
  const int lane = tid & 31;
  const int wv   = tid >> 5;
  const long n0  = (long)blockIdx.x * 16;

  for (int idx = tid; idx < 16 * 128; idx += 128) {
    int mm = idx >> 7, k = idx & 127;
    float v = (k < HID) ? x_buf[(n0 + mm) * HID + k]
                        : agg_s[(n0 + mm) * HID + (k - HID)];
    sCat[idx] = f2bf(v);
  }
  for (int idx = tid; idx < 16 * 192; idx += 128) {
    int mm = idx / 192, k = idx % 192;
    sV[mm * 192 + k] = f2bf(v_buf[(n0 + mm) * 192 + k]);
  }
  __syncthreads();

  const int ncol = wv * 16 + (lane & 15);
  const int mh   = (lane < 16) ? 0 : 8;
  const int m    = lane & 15;
  const int kaOff = (lane < 16) ? 0 : 8;
  const int kbOff = (lane < 16) ? 0 : 8;

  // u = silu([x, agg_s] @ uw1 + b1) ; K = 128
  v8f acc = {};
  for (int kk = 0; kk < 4; ++kk) {
    AFrag a, b;
    int ka = kk * 32 + kaOff;
    a.h[0] = *(const bf16x8*)(&sCat[m * 128 + ka]);
    a.h[1] = *(const bf16x8*)(&sCat[m * 128 + ka + 16]);
    const __bf16* bp = uw1t + (long)ncol * 128 + kk * 32 + kbOff;
    b.h[0] = *(const bf16x8*)(bp);
    b.h[1] = *(const bf16x8*)(bp + 16);
    acc = __builtin_amdgcn_wmma_f32_16x16x32_bf16(false, a.v, false, b.v,
                                                  (short)0, acc, false, false);
  }
  {
    float bias = ub1[ncol];
    #pragma unroll
    for (int r = 0; r < 8; ++r) {
      int mm = mh + r;
      sU[mm * 64 + ncol] = f2bf(silu(acc[r] + bias));
    }
  }
  __syncthreads();

  // x += u @ uw2 + b2 ; K = 64
  v8f accx = {};
  for (int kk = 0; kk < 2; ++kk) {
    AFrag a, b;
    int ka = kk * 32 + kaOff;
    a.h[0] = *(const bf16x8*)(&sU[m * 64 + ka]);
    a.h[1] = *(const bf16x8*)(&sU[m * 64 + ka + 16]);
    const __bf16* bp = uw2t + (long)ncol * 64 + kk * 32 + kbOff;
    b.h[0] = *(const bf16x8*)(bp);
    b.h[1] = *(const bf16x8*)(bp + 16);
    accx = __builtin_amdgcn_wmma_f32_16x16x32_bf16(false, a.v, false, b.v,
                                                   (short)0, accx, false, false);
  }
  {
    float bias2 = ub2[ncol];
    #pragma unroll
    for (int r = 0; r < 8; ++r) {
      int mm = mh + r;
      long n = n0 + mm;
      float xv = x_buf[n * HID + ncol] + accx[r] + bias2;
      x_buf[n * HID + ncol] = xv;
      xs_out[n * HID + ncol] = xv;
    }
  }

  // v = v + agg_v + v @ mix  (3 independent 16x64x64 GEMMs)
  for (int i = 0; i < 3; ++i) {
    v8f accv = {};
    for (int kk = 0; kk < 2; ++kk) {
      AFrag a, b;
      int ka = kk * 32 + kaOff;
      a.h[0] = *(const bf16x8*)(&sV[m * 192 + i * 64 + ka]);
      a.h[1] = *(const bf16x8*)(&sV[m * 192 + i * 64 + ka + 16]);
      const __bf16* bp = mixt + (long)ncol * 64 + kk * 32 + kbOff;
      b.h[0] = *(const bf16x8*)(bp);
      b.h[1] = *(const bf16x8*)(bp + 16);
      accv = __builtin_amdgcn_wmma_f32_16x16x32_bf16(false, a.v, false, b.v,
                                                     (short)0, accv, false, false);
    }
    #pragma unroll
    for (int r = 0; r < 8; ++r) {
      int mm = mh + r;
      long off = (n0 + mm) * 192 + i * 64 + ncol;
      float vv = v_buf[off] + agg_v[off] + accv[r];
      v_buf[off] = vv;
      vs_out[off] = vv;
    }
  }
}

// ---------------- readout ----------------
__global__ void straw_zero_dip(float* __restrict__ dip) {
  int i = blockIdx.x * blockDim.x + threadIdx.x;
  if (i < N_GRAPHS * 3) dip[i] = 0.0f;
}

__global__ __launch_bounds__(96) void straw_readout(
    const float* __restrict__ v_buf, const int* __restrict__ batch,
    const float* __restrict__ fw, float* __restrict__ dip) {
  long n = blockIdx.x;
  int g = threadIdx.x >> 5, lane = threadIdx.x & 31;
  float s = 0.0f;
  for (int h = lane; h < HID; h += 32)
    s += v_buf[n * 192 + g * 64 + h] * fw[h];
  #pragma unroll
  for (int off = 16; off > 0; off >>= 1) s += __shfl_down(s, off, 32);
  if (lane == 0) atomicAdd(&dip[(long)batch[n] * 3 + g], s);
}

// ---------------- host launcher ----------------
extern "C" void kernel_launch(void* const* d_in, const int* in_sizes, int n_in,
                              void* d_out, int out_size, void* d_ws, size_t ws_size,
                              hipStream_t stream) {
  const int*   z      = (const int*)  d_in[0];
  const float* pos    = (const float*)d_in[1];
  const int*   eidx   = (const int*)  d_in[2];
  const int*   batch  = (const int*)  d_in[3];
  const float* emb    = (const float*)d_in[4];
  const float* msg_w1 = (const float*)d_in[5];
  const float* msg_b1 = (const float*)d_in[6];
  const float* msg_w2 = (const float*)d_in[7];
  const float* msg_b2 = (const float*)d_in[8];
  const float* upd_w1 = (const float*)d_in[9];
  const float* upd_b1 = (const float*)d_in[10];
  const float* upd_w2 = (const float*)d_in[11];
  const float* upd_b2 = (const float*)d_in[12];
  const float* mix_w  = (const float*)d_in[13];
  const float* fw     = (const float*)d_in[14];
  (void)in_sizes; (void)n_in; (void)out_size; (void)ws_size;

  const int* row = eidx;
  const int* col = eidx + N_EDGES;

  // workspace carve-up (256B aligned)
  char* w = (char*)d_ws;
  size_t off = 0;
  auto carve = [&](size_t bytes) -> char* {
    char* p = w + off;
    off += (bytes + 255) & ~(size_t)255;
    return p;
  };
  float*  x_buf    = (float*) carve((size_t)N_NODES * HID * 4);
  float*  v_buf    = (float*) carve((size_t)N_NODES * 3 * HID * 4);
  float*  agg_s    = (float*) carve((size_t)N_NODES * HID * 4);
  float*  agg_v    = (float*) carve((size_t)N_NODES * 3 * HID * 4);
  float*  pseudo   = (float*) carve((size_t)N_NODES * 3 * 4);
  __bf16* nodefeat = (__bf16*)carve((size_t)N_NODES * NFEAT * 2);
  float*  w1last   = (float*) carve((size_t)L_LAYERS * NFEAT * 4);
  __bf16* W1T      = (__bf16*)carve((size_t)L_LAYERS * NFEAT * 512 * 2);
  __bf16* W2T      = (__bf16*)carve((size_t)L_LAYERS * NFEAT * NFEAT * 2);
  __bf16* UW1T     = (__bf16*)carve((size_t)L_LAYERS * HID * 128 * 2);
  __bf16* UW2T     = (__bf16*)carve((size_t)L_LAYERS * HID * HID * 2);
  __bf16* MIXT     = (__bf16*)carve((size_t)L_LAYERS * HID * HID * 2);

  // output layout: dip (128*3) ++ xs (3*N*64) ++ vs (3*N*192)
  float* dip = (float*)d_out;
  float* xs0 = dip + N_GRAPHS * 3;
  float* vs0 = xs0 + (size_t)L_LAYERS * N_NODES * HID;

  // ---- weight prep ----
  for (int l = 0; l < L_LAYERS; ++l) {
    straw_transpose_bf16<<<(512 * NFEAT + 255) / 256, 256, 0, stream>>>(
        msg_w1 + (size_t)l * 513 * NFEAT, W1T + (size_t)l * NFEAT * 512, 512, NFEAT);
    straw_transpose_bf16<<<(NFEAT * NFEAT + 255) / 256, 256, 0, stream>>>(
        msg_w2 + (size_t)l * NFEAT * NFEAT, W2T + (size_t)l * NFEAT * NFEAT, NFEAT, NFEAT);
    straw_transpose_bf16<<<(128 * HID + 255) / 256, 256, 0, stream>>>(
        upd_w1 + (size_t)l * 128 * HID, UW1T + (size_t)l * HID * 128, 128, HID);
    straw_transpose_bf16<<<(HID * HID + 255) / 256, 256, 0, stream>>>(
        upd_w2 + (size_t)l * HID * HID, UW2T + (size_t)l * HID * HID, HID, HID);
    straw_transpose_bf16<<<(HID * HID + 255) / 256, 256, 0, stream>>>(
        mix_w + (size_t)l * HID * HID, MIXT + (size_t)l * HID * HID, HID, HID);
  }
  straw_w1last<<<(L_LAYERS * NFEAT + 255) / 256, 256, 0, stream>>>(msg_w1, w1last);

  // ---- node init + dip zero ----
  straw_init_nodes<<<N_NODES, 256, 0, stream>>>(z, pos, emb, x_buf, v_buf);
  straw_zero_dip<<<2, 256, 0, stream>>>(dip);

  // ---- layers ----
  for (int l = 0; l < L_LAYERS; ++l) {
    straw_node_prep<<<N_NODES, 256, 0, stream>>>(x_buf, v_buf, nodefeat, pseudo,
                                                 agg_s, agg_v);
    straw_edge_mlp<<<N_EDGES / 16, 512, 0, stream>>>(
        nodefeat, pseudo, row, col,
        W1T + (size_t)l * NFEAT * 512, w1last + (size_t)l * NFEAT,
        msg_b1 + (size_t)l * NFEAT,
        W2T + (size_t)l * NFEAT * NFEAT, msg_b2 + (size_t)l * NFEAT,
        agg_s, agg_v);
    straw_node_update<<<N_NODES / 16, 128, 0, stream>>>(
        x_buf, v_buf, agg_s, agg_v,
        UW1T + (size_t)l * HID * 128, upd_b1 + (size_t)l * HID,
        UW2T + (size_t)l * HID * HID, upd_b2 + (size_t)l * HID,
        MIXT + (size_t)l * HID * HID,
        xs0 + (size_t)l * N_NODES * HID,
        vs0 + (size_t)l * N_NODES * 3 * HID);
  }

  // ---- readout ----
  straw_readout<<<N_NODES, 96, 0, stream>>>(v_buf, batch, fw, dip);
}
